// TitansMemoryModule_30013231465060
// MI455X (gfx1250) — matrix operand order repaired and checked by reading, not compiled, not run on hardware
//
#include <hip/hip_runtime.h>

// Problem constants (match reference)
#define Hdim   2048
#define Mslots 256
#define Bbatch 8
#define Sseq   4096
#define ROWS   (Bbatch * Sseq)   // 32768
#define TILE   16
#define NTILES (ROWS / TILE)     // 2048

// d_out layout (floats): combined [B,S,2H] | read_memory [B,S,H] | new_memory [B,M,H]
static constexpr size_t OUT_COMBINED = 0;
static constexpr size_t OUT_READ     = (size_t)ROWS * 2 * Hdim;            // 134217728
static constexpr size_t OUT_NEWMEM   = OUT_READ + (size_t)ROWS * Hdim;     // 201326592

// workspace layout (bytes)
static constexpr size_t WS_MEMB = 0;                                   // bf16 [256][2048]  1MB
static constexpr size_t WS_MEMT = 1u << 20;                            // bf16 [2048][256]  1MB
static constexpr size_t WS_TSUM = 2u << 20;                            // f32 [2048][2048] 16MB
static constexpr size_t WS_MEAN = WS_TSUM + (size_t)NTILES * Hdim * 4; // f32 [8][2048]
static constexpr size_t WS_WC   = WS_MEAN + (size_t)Bbatch * Hdim * 4; // f32 [8][2048]

typedef __attribute__((ext_vector_type(16))) __bf16       v16bf;
typedef __attribute__((ext_vector_type(8)))  float        v8f;
typedef __attribute__((ext_vector_type(4)))  unsigned int u32x4;
typedef __attribute__((ext_vector_type(4)))  float        f32x4;

union ABfrag { v16bf v; u32x4 u[2]; };

__device__ __forceinline__ unsigned int bf_bits(float f) {
  unsigned int u = __float_as_uint(f);
  return (u + 0x7FFFu + ((u >> 16) & 1u)) >> 16;   // round-to-nearest-even
}
__device__ __forceinline__ unsigned int pack2bf(float lo, float hi) {
  return bf_bits(lo) | (bf_bits(hi) << 16);
}

// ---------------------------------------------------------------------------
// Kernel 1: memory (fp32) -> bf16 row-major + bf16 transposed (both L2-resident)
// ---------------------------------------------------------------------------
__global__ void k_prep_mem(const float* __restrict__ mem,
                           unsigned short* __restrict__ memB,
                           unsigned short* __restrict__ memT) {
  const int m  = blockIdx.x;            // 0..255
  const int c0 = threadIdx.x * 8;       // 0..2040
  f32x4 a = *(const f32x4*)(mem + (size_t)m * Hdim + c0);
  f32x4 b = *(const f32x4*)(mem + (size_t)m * Hdim + c0 + 4);
  u32x4 p;
  p.x = pack2bf(a.x, a.y); p.y = pack2bf(a.z, a.w);
  p.z = pack2bf(b.x, b.y); p.w = pack2bf(b.z, b.w);
  *(u32x4*)(memB + (size_t)m * Hdim + c0) = p;
  float v[8] = {a.x, a.y, a.z, a.w, b.x, b.y, b.z, b.w};
#pragma unroll
  for (int j = 0; j < 8; ++j)
    memT[(size_t)(c0 + j) * Mslots + m] = (unsigned short)bf_bits(v[j]);
}

// ---------------------------------------------------------------------------
// Kernel 2 (main): per 16-row tile of x:
//   scores = x @ mem^T  (WMMA bf16)  -> softmax -> read = w @ mem (WMMA bf16)
//   emits combined (x | read), read_memory, and deterministic tile column-sums
// ---------------------------------------------------------------------------
__global__ void __launch_bounds__(256, 2)
k_fused_read(const float* __restrict__ x,
             const unsigned short* __restrict__ memB,
             const unsigned short* __restrict__ memT,
             float* __restrict__ out,
             float* __restrict__ tsum) {
  __shared__ union {
    unsigned short xs[TILE][Hdim];            // 64KB: bf16 x tile (phase 1)
    struct {
      float          sc[TILE][Mslots];        // 16KB: scores (phase 2)
      unsigned short w[TILE][Mslots];         //  8KB: softmax weights bf16
      float          red[TILE][16];           //  1KB: row reduction scratch
    } p2;
  } L;

  const int    t    = threadIdx.x;            // 0..255
  const int    tile = blockIdx.x;             // 0..2047
  const size_t r0   = (size_t)tile * TILE;

  // ---- Phase 1: load x tile, write combined-left, build bf16 LDS copy, col sums
  {
    const int c0 = t * 8;                     // 8 contiguous columns per thread
    float cs[8] = {0, 0, 0, 0, 0, 0, 0, 0};
#pragma unroll 4
    for (int rr = 0; rr < TILE; ++rr) {
      const float* xr = x + (r0 + rr) * (size_t)Hdim + c0;
      f32x4 a = *(const f32x4*)xr;
      f32x4 b = *(const f32x4*)(xr + 4);
      float* cb = out + OUT_COMBINED + (r0 + rr) * (size_t)(2 * Hdim) + c0;
      *(f32x4*)cb = a; *(f32x4*)(cb + 4) = b;
      cs[0] += a.x; cs[1] += a.y; cs[2] += a.z; cs[3] += a.w;
      cs[4] += b.x; cs[5] += b.y; cs[6] += b.z; cs[7] += b.w;
      u32x4 p;
      p.x = pack2bf(a.x, a.y); p.y = pack2bf(a.z, a.w);
      p.z = pack2bf(b.x, b.y); p.w = pack2bf(b.z, b.w);
      *(u32x4*)&L.xs[rr][c0] = p;
    }
    float* ts = tsum + (size_t)tile * Hdim + c0;
    f32x4 s0 = {cs[0], cs[1], cs[2], cs[3]};
    f32x4 s1 = {cs[4], cs[5], cs[6], cs[7]};
    *(f32x4*)ts = s0; *(f32x4*)(ts + 4) = s1;
  }
  __syncthreads();

  const int wave = t >> 5;
  const int lane = t & 31;
  const int lm   = lane & 15;     // row (A) or column (B/D)
  const int lh   = lane >> 4;     // lane-half selects K sub-range / M+8

  // ---- Phase 2: GEMM1  scores[16][256] = xtile(16xH) @ memB^T
  v8f acc0 = {}; v8f acc1 = {};
  {
    const int n0 = wave * 32;     // this wave's 32 memory slots
    for (int k0 = 0; k0 < Hdim; k0 += 32) {
      ABfrag af, b0, b1;
      af.u[0] = *(const u32x4*)&L.xs[lm][k0 + 8 * lh];
      af.u[1] = *(const u32x4*)&L.xs[lm][k0 + 16 + 8 * lh];
      const unsigned short* bp0 = memB + (size_t)(n0 + lm) * Hdim + k0 + 8 * lh;
      const unsigned short* bp1 = memB + (size_t)(n0 + 16 + lm) * Hdim + k0 + 8 * lh;
      b0.u[0] = *(const u32x4*)bp0; b0.u[1] = *(const u32x4*)(bp0 + 16);
      b1.u[0] = *(const u32x4*)bp1; b1.u[1] = *(const u32x4*)(bp1 + 16);
      acc0 = __builtin_amdgcn_wmma_f32_16x16x32_bf16(false, af.v, false, b0.v,
                                                     (short)0, acc0, false, false);
      acc1 = __builtin_amdgcn_wmma_f32_16x16x32_bf16(false, af.v, false, b1.v,
                                                     (short)0, acc1, false, false);
    }
  }
  __syncthreads();                               // xs dead; safe to alias with p2
  {
    const int n0 = wave * 32;
#pragma unroll
    for (int v = 0; v < 8; ++v) {
      L.p2.sc[v + 8 * lh][n0 + lm]      = acc0[v];
      L.p2.sc[v + 8 * lh][n0 + 16 + lm] = acc1[v];
    }
  }
  __syncthreads();

  // ---- Phase 3: softmax over M=256 (row = t>>4, 16-element chunk = t&15)
  {
    const int row = t >> 4, g = t & 15;
    const float* srow = &L.p2.sc[row][0];
    float mx = srow[g * 16];
#pragma unroll
    for (int j = 1; j < 16; ++j) mx = fmaxf(mx, srow[g * 16 + j]);
    L.p2.red[row][g] = mx;
    __syncthreads();
    float rmax = L.p2.red[row][0];
#pragma unroll
    for (int j = 1; j < 16; ++j) rmax = fmaxf(rmax, L.p2.red[row][j]);
    __syncthreads();
    float e[16]; float ls = 0.f;
#pragma unroll
    for (int j = 0; j < 16; ++j) { e[j] = __expf(srow[g * 16 + j] - rmax); ls += e[j]; }
    L.p2.red[row][g] = ls;
    __syncthreads();
    float rsum = 0.f;
#pragma unroll
    for (int j = 0; j < 16; ++j) rsum += L.p2.red[row][j];
    const float inv = 1.0f / rsum;
    u32x4 p0, p1;
    p0.x = pack2bf(e[0] * inv,  e[1] * inv);  p0.y = pack2bf(e[2] * inv,  e[3] * inv);
    p0.z = pack2bf(e[4] * inv,  e[5] * inv);  p0.w = pack2bf(e[6] * inv,  e[7] * inv);
    p1.x = pack2bf(e[8] * inv,  e[9] * inv);  p1.y = pack2bf(e[10] * inv, e[11] * inv);
    p1.z = pack2bf(e[12] * inv, e[13] * inv); p1.w = pack2bf(e[14] * inv, e[15] * inv);
    *(u32x4*)&L.p2.w[row][g * 16]     = p0;
    *(u32x4*)&L.p2.w[row][g * 16 + 8] = p1;
  }
  __syncthreads();

  // ---- Phase 4: GEMM2  read[16][2048] = w(16x256) @ memT^T ; dual-destination stores
  for (int j = 0; j < 16; ++j) {
    const int h0 = wave * 256 + j * 16;        // this wave covers h in [256w, 256w+256)
    v8f acc = {};
#pragma unroll
    for (int k0 = 0; k0 < Mslots; k0 += 32) {
      ABfrag af, bf;
      af.u[0] = *(const u32x4*)&L.p2.w[lm][k0 + 8 * lh];
      af.u[1] = *(const u32x4*)&L.p2.w[lm][k0 + 16 + 8 * lh];
      const unsigned short* bp = memT + (size_t)(h0 + lm) * Mslots + k0 + 8 * lh;
      bf.u[0] = *(const u32x4*)bp; bf.u[1] = *(const u32x4*)(bp + 16);
      acc = __builtin_amdgcn_wmma_f32_16x16x32_bf16(false, af.v, false, bf.v,
                                                    (short)0, acc, false, false);
    }
#pragma unroll
    for (int v = 0; v < 8; ++v) {
      const int    row = v + 8 * lh;
      const size_t r   = r0 + row;
      const float  val = acc[v];
      out[OUT_COMBINED + r * (size_t)(2 * Hdim) + Hdim + h0 + lm] = val;
      out[OUT_READ + r * (size_t)Hdim + h0 + lm]                  = val;
    }
  }
}

// ---------------------------------------------------------------------------
// Kernel 3: deterministic reduction of per-tile column sums -> mean over S
// ---------------------------------------------------------------------------
__global__ void k_reduce_mean(const float* __restrict__ tsum, float* __restrict__ mean) {
  const int b = blockIdx.x >> 3;
  const int h = ((blockIdx.x & 7) * 256) + threadIdx.x;
  const float* p = tsum + (size_t)(b * 256) * Hdim + h;   // 256 tiles per batch
  float s = 0.f;
  for (int i = 0; i < 256; ++i) s += p[(size_t)i * Hdim];
  mean[b * Hdim + h] = s * (1.0f / (float)Sseq);
}

// ---------------------------------------------------------------------------
// Kernel 4: write_content[b][i] = sum_h mean[b][h] * W[i][h] + bias[i]
// One block per 256 output columns; all 8 batch means staged in LDS so each
// W row is streamed from HBM exactly once (16MB total instead of 128MB).
// ---------------------------------------------------------------------------
__global__ void k_write_content(const float* __restrict__ mean,
                                const float* __restrict__ W,
                                const float* __restrict__ bias,
                                float* __restrict__ wc) {
  __shared__ float mall[Bbatch][Hdim];          // 64KB
  const int i = blockIdx.x * 256 + threadIdx.x; // output column
  for (int j = threadIdx.x; j < Bbatch * Hdim; j += 256)
    ((float*)mall)[j] = mean[j];
  __syncthreads();
  const float* wr = W + (size_t)i * Hdim;
  float acc[Bbatch] = {0, 0, 0, 0, 0, 0, 0, 0};
  for (int j = 0; j < Hdim; j += 4) {
    f32x4 wv = *(const f32x4*)(wr + j);
#pragma unroll
    for (int b = 0; b < Bbatch; ++b) {
      acc[b] += mall[b][j]     * wv.x + mall[b][j + 1] * wv.y +
                mall[b][j + 2] * wv.z + mall[b][j + 3] * wv.w;
    }
  }
  const float bi = bias[i];
#pragma unroll
  for (int b = 0; b < Bbatch; ++b) wc[b * Hdim + i] = acc[b] + bi;
}

// ---------------------------------------------------------------------------
// Kernel 5: new_memory[b,m,h] = 0.9*memory[m,h] + 0.1*write_content[b,h]
// ---------------------------------------------------------------------------
__global__ void k_new_memory(const float* __restrict__ mem,
                             const float* __restrict__ wc,
                             float* __restrict__ out) {
  const size_t idx = ((size_t)blockIdx.x * 256 + threadIdx.x) * 4;
  const int h = (int)(idx % Hdim);
  const size_t tmp = idx / Hdim;
  const int m = (int)(tmp % Mslots);
  const int b = (int)(tmp / Mslots);
  f32x4 mm = *(const f32x4*)(mem + (size_t)m * Hdim + h);
  f32x4 ww = *(const f32x4*)(wc + (size_t)b * Hdim + h);
  f32x4 r  = mm * 0.9f + ww * 0.1f;
  *(f32x4*)(out + OUT_NEWMEM + idx) = r;
}

// ---------------------------------------------------------------------------
extern "C" void kernel_launch(void* const* d_in, const int* in_sizes, int n_in,
                              void* d_out, int out_size, void* d_ws, size_t ws_size,
                              hipStream_t stream) {
  const float* x       = (const float*)d_in[0];
  const float* memory  = (const float*)d_in[1];
  const float* w_write = (const float*)d_in[2];
  const float* b_write = (const float*)d_in[3];
  float* out = (float*)d_out;

  char* ws = (char*)d_ws;
  unsigned short* memB = (unsigned short*)(ws + WS_MEMB);
  unsigned short* memT = (unsigned short*)(ws + WS_MEMT);
  float*          tsum = (float*)(ws + WS_TSUM);
  float*          mean = (float*)(ws + WS_MEAN);
  float*          wc   = (float*)(ws + WS_WC);

  k_prep_mem<<<Mslots, 256, 0, stream>>>(memory, memB, memT);
  k_fused_read<<<NTILES, 256, 0, stream>>>(x, memB, memT, out, tsum);
  k_reduce_mean<<<64, 256, 0, stream>>>(tsum, mean);
  k_write_content<<<Hdim / 256, 256, 0, stream>>>(mean, w_write, b_write, wc);
  k_new_memory<<<(Bbatch * Mslots * Hdim) / (256 * 4), 256, 0, stream>>>(memory, wc, out);
}
